// VectorQuantizer_67637144978418
// MI455X (gfx1250) — compile-verified, matrix operand order checked by reading, not run on hardware
//
#include <hip/hip_runtime.h>

typedef __attribute__((ext_vector_type(16))) __bf16 v16bf;
typedef __attribute__((ext_vector_type(8))) float v8f;

#define DDIM 64
#define KDIM 512
#define NROWS 65536
#define ROWS_PER_WAVE 16
#define WAVES_PER_BLOCK 8
#define ROWS_PER_BLOCK (ROWS_PER_WAVE * WAVES_PER_BLOCK) /* 128 */
#define NBLOCKS (NROWS / ROWS_PER_BLOCK)                 /* 512 */

#define TILES_PER_STAGE 8                 /* 8 column-tiles of 16 codes */
#define NSTAGES (KDIM / (TILES_PER_STAGE * 16)) /* 4 */
/* per stage: 8 tiles * 2 K-chunks * 2 (hi/lo) * 32 lanes * 16 bf16 = 16384 */
#define STAGE_ELEMS (TILES_PER_STAGE * 2 * 2 * 32 * 16)

// Main kernel: distances via error-compensated bf16 WMMA (bf16x3 split),
// per-row argmin, gather, STE output, per-block loss partial.
__global__ __launch_bounds__(256) void vq_main(const float* __restrict__ x,
                                               const float* __restrict__ emb,
                                               float* __restrict__ out,
                                               float* __restrict__ partial) {
    __shared__ __align__(128) __bf16 s_bfrag[STAGE_ELEMS]; /* 32 KB */
    __shared__ float s_enorm[KDIM];
    __shared__ int s_idx[ROWS_PER_BLOCK];
    __shared__ float s_red[256];

    const int tid = threadIdx.x;
    const int lane = tid & 31;
    const int wave = tid >> 5;
    const int half = lane >> 4; /* which 16-lane half of the wave32 */
    const int l16 = lane & 15;

    // Codebook squared norms ||e_k||^2 -> LDS, exact fp32 (once per block).
    for (int k = tid; k < KDIM; k += 256) {
        float s = 0.f;
#pragma unroll 8
        for (int d = 0; d < DDIM; ++d) {
            float e = emb[d * KDIM + k];
            s += e * e;
        }
        s_enorm[k] = s;
    }

    const int rowBase = blockIdx.x * ROWS_PER_BLOCK + wave * ROWS_PER_WAVE;

    // ---- A fragments (bf16 hi/lo split), ISA 16-bit A 16x32 layout ----
    // Lane l16 = row M. For chunk c (K-range c*32..c*32+31), vector element e
    // holds K = c*32 + (e>>3)*16 + half*8 + (e&7).
    v16bf aH[2], aL[2];
    {
        const float* xr = x + (size_t)(rowBase + l16) * DDIM;
#pragma unroll
        for (int c = 0; c < 2; ++c) {
#pragma unroll
            for (int e = 0; e < 16; ++e) {
                int k = c * 32 + ((e >> 3) << 4) + (half << 3) + (e & 7);
                float v = xr[k];
                __bf16 h = (__bf16)v;
                aH[c][e] = h;
                aL[c][e] = (__bf16)(v - (float)h);
            }
        }
    }

    float bestV[8];
    int bestI[8];
#pragma unroll
    for (int j = 0; j < 8; ++j) {
        bestV[j] = 3.4e38f;
        bestI[j] = 0;
    }

    for (int stage = 0; stage < NSTAGES; ++stage) {
        const int tileBase = stage * TILES_PER_STAGE; /* global tile index */
        __syncthreads(); /* previous stage fully consumed (and s_enorm ready) */

        // ---- Stage codebook fragments into LDS in wave-operand order ----
        // Position p encodes (t, c, lane_p, e); operand slot layout:
        //   op = (t*2 + c)*2 + m   (m: 0=hi, 1=lo), bf16 idx = op*512 + lane*16 + e.
        // Value: col = (tileBase+t)*16 + (lane_p&15),
        //        K   = c*32 + (e>>3)*16 + (lane_p>>4)*8 + (e&7).
        for (int p = tid; p < STAGE_ELEMS / 2; p += 256) {
            int e = p & 15;
            int lane_p = (p >> 4) & 31;
            int c = (p >> 9) & 1;
            int t = p >> 10;
            int col = (tileBase + t) * 16 + (lane_p & 15);
            int k = c * 32 + ((e >> 3) << 4) + ((lane_p >> 4) << 3) + (e & 7);
            float v = emb[(size_t)k * KDIM + col];
            __bf16 h = (__bf16)v;
            int base = ((t * 2 + c) * 2) * 512 + lane_p * 16 + e;
            s_bfrag[base] = h;                          /* hi operand */
            s_bfrag[base + 512] = (__bf16)(v - (float)h); /* lo operand */
        }
        __syncthreads();

        // ---- Process 8 tiles from LDS: 8 ds_loads + 6 WMMAs per tile ----
        for (int t = 0; t < TILES_PER_STAGE; ++t) {
            const __bf16* tb = s_bfrag + (t * 4) * 512 + lane * 16;
            v16bf bh0 = *(const v16bf*)(tb);
            v16bf bl0 = *(const v16bf*)(tb + 512);
            v16bf bh1 = *(const v16bf*)(tb + 1024);
            v16bf bl1 = *(const v16bf*)(tb + 1536);

            v8f c = {};
            // S ~= Ahi*Bhi + Ahi*Blo + Alo*Bhi, fp32 accumulate (per K-chunk)
            c = __builtin_amdgcn_wmma_f32_16x16x32_bf16(false, aH[0], false, bh0,
                                                        (short)0, c, false, false);
            c = __builtin_amdgcn_wmma_f32_16x16x32_bf16(false, aH[0], false, bl0,
                                                        (short)0, c, false, false);
            c = __builtin_amdgcn_wmma_f32_16x16x32_bf16(false, aL[0], false, bh0,
                                                        (short)0, c, false, false);
            c = __builtin_amdgcn_wmma_f32_16x16x32_bf16(false, aH[1], false, bh1,
                                                        (short)0, c, false, false);
            c = __builtin_amdgcn_wmma_f32_16x16x32_bf16(false, aH[1], false, bl1,
                                                        (short)0, c, false, false);
            c = __builtin_amdgcn_wmma_f32_16x16x32_bf16(false, aL[1], false, bh1,
                                                        (short)0, c, false, false);

            // C layout: lane l16 = N, VGPR j = M (lanes 0-15) / M+8 (16-31).
            const int kcol = (tileBase + t) * 16 + l16;
            const float en = s_enorm[kcol];
#pragma unroll
            for (int j = 0; j < 8; ++j) {
                float dist = en - 2.0f * c[j];
                if (dist < bestV[j]) { /* strict < keeps smallest index on ties */
                    bestV[j] = dist;
                    bestI[j] = kcol;
                }
            }
        }
    }

    // Argmin across the 16 columns held by each half of the wave.
#pragma unroll
    for (int off = 1; off < 16; off <<= 1) {
#pragma unroll
        for (int j = 0; j < 8; ++j) {
            float ov = __shfl_xor(bestV[j], off, 32);
            int oi = __shfl_xor(bestI[j], off, 32);
            if (ov < bestV[j] || (ov == bestV[j] && oi < bestI[j])) {
                bestV[j] = ov;
                bestI[j] = oi;
            }
        }
    }
    if (l16 == 0) {
#pragma unroll
        for (int j = 0; j < 8; ++j)
            s_idx[wave * 16 + half * 8 + j] = bestI[j]; /* rows M=j / M=j+8 */
    }
    __syncthreads();

    // Gather codebook column (exact fp32), straight-through output, loss.
    float lsum = 0.f;
#pragma unroll 4
    for (int r = 0; r < ROWS_PER_WAVE; ++r) {
        const int row = rowBase + r;
        const int kidx = s_idx[wave * 16 + r];
        const float* xp = x + (size_t)row * DDIM;
        float* op = out + (size_t)row * DDIM;
        float q0 = emb[(size_t)lane * KDIM + kidx];
        float q1 = emb[(size_t)(lane + 32) * KDIM + kidx];
        float x0 = xp[lane];
        float x1 = xp[lane + 32];
        float r0 = q0 - x0;
        float r1 = q1 - x1;
        op[lane] = x0 + r0; /* x + stop_grad(q - x), reference rounding */
        op[lane + 32] = x1 + r1;
        lsum += r0 * r0 + r1 * r1;
    }

    // Deterministic block reduction of the loss partial.
    s_red[tid] = lsum;
    __syncthreads();
    for (int s = 128; s > 0; s >>= 1) {
        if (tid < s) s_red[tid] += s_red[tid + s];
        __syncthreads();
    }
    if (tid == 0) partial[blockIdx.x] = s_red[0];
}

// Final fixed-order reduction of the 512 block partials -> scalar loss.
__global__ __launch_bounds__(256) void vq_loss(const float* __restrict__ partial,
                                               float* __restrict__ loss_out) {
    __shared__ float s_red[256];
    const int tid = threadIdx.x;
    s_red[tid] = partial[tid] + partial[tid + 256];
    __syncthreads();
    for (int t = 128; t > 0; t >>= 1) {
        if (tid < t) s_red[tid] += s_red[tid + t];
        __syncthreads();
    }
    if (tid == 0)
        loss_out[0] = 1.25f * s_red[0] / (float)(NROWS * DDIM);
}

extern "C" void kernel_launch(void* const* d_in, const int* in_sizes, int n_in,
                              void* d_out, int out_size, void* d_ws, size_t ws_size,
                              hipStream_t stream) {
    const float* x = (const float*)d_in[0];   /* [64,32,32,64] -> [65536,64] */
    const float* emb = (const float*)d_in[1]; /* [64,512] */
    float* out = (float*)d_out;               /* 65536*64 outputs + 1 loss */
    float* partial = (float*)d_ws;            /* 512 floats of scratch */

    vq_main<<<NBLOCKS, 256, 0, stream>>>(x, emb, out, partial);
    vq_loss<<<1, 256, 0, stream>>>(partial, out + (size_t)NROWS * DDIM);
}
